// RelativeLoss_53025666237022
// MI455X (gfx1250) — matrix operand order repaired and checked by reading, not compiled, not run on hardware
//
#include <hip/hip_runtime.h>
#include <hip/hip_fp16.h>
#include <math.h>

// ---------------------------------------------------------------------------
// RelativeLoss fused batched-Gram kernel for gfx1250 (MI455X, wave32, WMMA).
//
//   bs=32, nsl=1024, hs=768; cos = Xn @ Xn^T per batch (51.5 GFLOP), fused
//   masked-loss epilogue; the 1024x1024 cosine matrices are never stored.
//
// Structure:
//   * 128x128 macro-tile per 256-thread block (8 waves); wave w owns a
//     16-row A strip and all 8 j-tiles (8 v8f accumulators).
//   * B block staged to LDS via global_load_async_to_lds_b128 (ASYNCcnt),
//     double-buffered: staging K-step t+1 overlaps the WMMAs of step t.
//   * Round-4 change: sched_group_barrier pins the issue order per K-step to
//       2x VMEM-read (A) -> 16x DS-read (all B fragments) -> 8x WMMA
//     so RA keeps 16 ds_load_b128 results live and the compiler emits
//     graduated s_wait_dscnt instead of the ds_load -> wait 0 -> wmma
//     serial chain the default scheduler produced.
//   * Global (L2) traffic: 24 B per output cell; xn (48 MB) is L2-resident.
// ---------------------------------------------------------------------------

typedef __attribute__((ext_vector_type(16))) _Float16 v16h;
typedef __attribute__((ext_vector_type(4)))  _Float16 v4h;
typedef __attribute__((ext_vector_type(8)))  float    v8f;

#define BS  32
#define NSL 1024
#define HS  768

#define LDS_ROW   80                 // 64B of B data + 16B pad (bank spread)
#define LDS_BUF   (128 * LDS_ROW)    // one K-step of B: 128 rows x 32 halves

// sched_group_barrier masks (LLVM AMDGPU)
#define SG_WMMA      0x008
#define SG_VMEM_READ 0x020
#define SG_DS_READ   0x100

struct HalfChunks { uint4 lo; uint4 hi; };   // 32B == v16h

// --- Kernel 1: row L2-normalize f32 -> f16, one wave per row; zero accums ---
__global__ __launch_bounds__(256) void rl_normalize_kernel(
    const float* __restrict__ embeds, _Float16* __restrict__ xn,
    float* __restrict__ numAcc) {
  if (blockIdx.x == 0 && threadIdx.x < BS) numAcc[threadIdx.x] = 0.0f;

  const int wave = threadIdx.x >> 5;
  const int lane = threadIdx.x & 31;
  const long long row = (long long)blockIdx.x * 8 + wave;   // 32768 rows
  const float* __restrict__ src = embeds + row * HS;
  _Float16* __restrict__ dst = xn + row * HS;

  float4 v[6];
  float ss = 0.0f;
#pragma unroll
  for (int w = 0; w < 6; ++w) {                 // 6 * 32 lanes * 4 = 768
    v[w] = ((const float4*)src)[lane + w * 32];
    ss += v[w].x * v[w].x + v[w].y * v[w].y + v[w].z * v[w].z + v[w].w * v[w].w;
  }
#pragma unroll
  for (int off = 16; off > 0; off >>= 1) ss += __shfl_xor(ss, off, 32);

  const float scale = 1.0f / fmaxf(sqrtf(ss), 1e-8f);
#pragma unroll
  for (int w = 0; w < 6; ++w) {
    v4h h;
    h.x = (_Float16)(v[w].x * scale);
    h.y = (_Float16)(v[w].y * scale);
    h.z = (_Float16)(v[w].z * scale);
    h.w = (_Float16)(v[w].w * scale);
    ((v4h*)dst)[lane + w * 32] = h;             // 8B coalesced stores
  }
}

// Async-stage one 16B chunk of the B block (row-major, padded rows) into LDS.
// chunk c (0..511): row = c>>2 (j within macro-tile), 16B piece = c&3.
__device__ __forceinline__ void rl_stage_chunk(
    const _Float16* __restrict__ base, int j0m, int kb, int c, unsigned ldsBase) {
  const _Float16* src = base + (long long)(j0m + (c >> 2)) * HS + kb + (c & 3) * 8;
  const unsigned lds = ldsBase + (unsigned)((c >> 2) * LDS_ROW + (c & 3) * 16);
  const unsigned long long ga = (unsigned long long)(uintptr_t)src;
  asm volatile("global_load_async_to_lds_b128 %0, %1, off"
               :: "v"(lds), "v"(ga) : "memory");
}

// --- Kernel 2: 128x128 macro-tile, LDS-staged B, 8 WMMA tiles per wave -----
__global__ __launch_bounds__(256) void rl_tile_loss_kernel(
    const _Float16* __restrict__ xn, const int* __restrict__ gt,
    const unsigned char* __restrict__ mask, float* __restrict__ numAcc) {
  __shared__ __align__(16) char Bsh[2][LDS_BUF];

  // Decode upper-triangular 8x8 macro-tile pair from blockIdx.x (36 pairs).
  int t = blockIdx.x, im = 0;
  while (t >= 8 - im) { t -= 8 - im; ++im; }
  const int jm = im + t;
  const int b  = blockIdx.y;

  const int tid  = threadIdx.x;
  const int wave = tid >> 5;
  const int lane = tid & 31;
  const int r    = lane & 15;
  const int hi   = lane >> 4;

  const _Float16* __restrict__ base = xn + (long long)b * NSL * HS;
  const int i0m = im * 128;                     // macro-tile row origin
  const int j0m = jm * 128;                     // macro-tile col origin

  // A layout (16-bit A 16x32): lane holds row i; halves 0..15
  // contiguous at K = kb + hi*16 (matches B).
  const _Float16* __restrict__ arow = base + (long long)(i0m + wave * 16 + r) * HS;

  const unsigned ldsB0 = (unsigned)(uintptr_t)&Bsh[0][0];
  const unsigned ldsB1 = ldsB0 + LDS_BUF;

  // Prologue: stage K-step 0 into buffer 0 (2 chunks per thread = 8KB/block).
  rl_stage_chunk(base, j0m, 0, tid,       ldsB0);
  rl_stage_chunk(base, j0m, 0, tid + 256, ldsB0);
  asm volatile("s_wait_asynccnt 0x0" ::: "memory");
  __syncthreads();

  v8f acc[8];
#pragma unroll
  for (int jt = 0; jt < 8; ++jt) acc[jt] = v8f{};

  int buf = 0;
  for (int kb = 0; kb < HS; kb += 32) {
    // Prefetch next K-step's B block into the other buffer (async).
    if (kb + 32 < HS) {
      const unsigned dst = buf ? ldsB0 : ldsB1;
      rl_stage_chunk(base, j0m, kb + 32, tid,       dst);
      rl_stage_chunk(base, j0m, kb + 32, tid + 256, dst);
    }

    // A fragment (global, private to this wave, reused across 8 j-tiles).
    HalfChunks af;
    af.lo = *(const uint4*)(arow + kb + hi * 16);
    af.hi = *(const uint4*)(arow + kb + hi * 16 + 8);
    const v16h av = __builtin_bit_cast(v16h, af);

    // B layout, same k order as A (16-bit B 32x16): lane holds column j0+r, halves 0..15
    // contiguous at K = kb + hi*16 -> 32B per lane from the staged row.
    const char* bufPtr = &Bsh[buf][0];
    v16h bvs[8];
#pragma unroll
    for (int jt = 0; jt < 8; ++jt) {
      const char* bb = bufPtr + (jt * 16 + r) * LDS_ROW + hi * 32;
      HalfChunks bf;
      bf.lo = *(const uint4*)bb;
      bf.hi = *(const uint4*)(bb + 16);
      bvs[jt] = __builtin_bit_cast(v16h, bf);
    }
#pragma unroll
    for (int jt = 0; jt < 8; ++jt) {
      acc[jt] = __builtin_amdgcn_wmma_f32_16x16x32_f16(
          false, av, false, bvs[jt], (short)0, acc[jt], false, false);
    }

    // Pin issue order for this region: A loads, then ALL 16 B-fragment DS
    // reads (kept live in distinct regs -> graduated s_wait_dscnt), then the
    // 8-WMMA chain. DS returns are in-order, so WMMA n waits only for its
    // own two loads while the rest stay in flight.
    __builtin_amdgcn_sched_group_barrier(SG_VMEM_READ, 2,  0);
    __builtin_amdgcn_sched_group_barrier(SG_DS_READ,   16, 0);
    __builtin_amdgcn_sched_group_barrier(SG_WMMA,      8,  0);

    // Own async stores to buf^1 done; everyone done reading buf -> flip.
    asm volatile("s_wait_asynccnt 0x0" ::: "memory");
    __syncthreads();
    buf ^= 1;
  }

  // Epilogue: C/D layout -> VGPR v, lane L: M = v + 8*(L>=16), N = L&15.
  const int itg = im * 8 + wave;
  float s = 0.0f;
#pragma unroll
  for (int jt = 0; jt < 8; ++jt) {
    const int jtg = jm * 8 + jt;
    if (jtg < itg) continue;                    // tile fully below diagonal
    const int   j  = jtg * 16 + r;
    const float gj = (float)gt[b * NSL + j];
    const bool  mj = mask[b * NSL + j] != 0;
#pragma unroll
    for (int v = 0; v < 8; ++v) {
      const int i = itg * 16 + v + hi * 8;
      if (j > i) {
        if (mj && (mask[b * NSL + i] != 0)) {
          const float gi = (float)gt[b * NSL + i];
          const float vd = 2.0f * fabsf(gi - gj) / (fabsf(gi) + fabsf(gj) + 1e-18f);
          const float d  = (1.0f - acc[jt][v]) - vd;
          s += d * d;
        }
      }
    }
  }
#pragma unroll
  for (int off = 16; off > 0; off >>= 1) s += __shfl_xor(s, off, 32);
  if (lane == 0) atomicAdd(&numAcc[b], s);
}

// --- Kernel 3: den_b = S_b*(S_b-1)/2; loss = mean_b sqrt(num/den) ----------
__global__ __launch_bounds__(32) void rl_finalize_kernel(
    const unsigned char* __restrict__ mask, const float* __restrict__ numAcc,
    float* __restrict__ out) {
  const int b = threadIdx.x;                    // 32 lanes, one per batch
  float S = 0.0f;
  for (int i = 0; i < NSL; ++i) S += (mask[b * NSL + i] != 0) ? 1.0f : 0.0f;
  const float den  = 0.5f * S * (S - 1.0f) + 1e-13f;
  float loss = sqrtf(numAcc[b] / den);
#pragma unroll
  for (int off = 16; off > 0; off >>= 1) loss += __shfl_xor(loss, off, 32);
  if (b == 0) out[0] = loss * (1.0f / (float)BS);
}

extern "C" void kernel_launch(void* const* d_in, const int* in_sizes, int n_in,
                              void* d_out, int out_size, void* d_ws, size_t ws_size,
                              hipStream_t stream) {
  const float*         embeds = (const float*)d_in[0];          // f32 [32,1024,768]
  const int*           gt     = (const int*)d_in[1];            // i32 [32,1024]
  const unsigned char* mask   = (const unsigned char*)d_in[2];  // bool [32,1024]

  _Float16* xn     = (_Float16*)d_ws;                           // 48 MiB (fits L2)
  float*    numAcc = (float*)((char*)d_ws +
                     (size_t)BS * NSL * HS * sizeof(_Float16)); // 32 floats
  float*    out    = (float*)d_out;

  // 1) normalize (8 rows/block * 4096 blocks = 32768 rows); zeroes numAcc
  rl_normalize_kernel<<<dim3(4096), dim3(256), 0, stream>>>(embeds, xn, numAcc);

  // 2) 36 upper-triangular 128x128 macro-tiles per batch, 32 batches
  rl_tile_loss_kernel<<<dim3(36, 32), dim3(256), 0, stream>>>(xn, gt, mask, numAcc);

  // 3) finalize scalar loss
  rl_finalize_kernel<<<dim3(1), dim3(32), 0, stream>>>(mask, numAcc, out);
}
